// BahdanauAttention_23914377904440
// MI455X (gfx1250) — matrix operand order repaired and checked by reading, not compiled, not run on hardware
//
#include <hip/hip_runtime.h>
#include <hip/hip_bf16.h>
#include <math.h>

// Problem constants
#define B_ 64
#define S_ 2048
#define E_ 512
#define D_ 512
#define A_ 256

typedef __attribute__((ext_vector_type(16))) __bf16 v16bf;
typedef __attribute__((ext_vector_type(4)))  __bf16 v4bf;
typedef __attribute__((ext_vector_type(8)))  float  v8f;

#if __has_builtin(__builtin_amdgcn_tanhf)
__device__ __forceinline__ float fast_tanh(float x) { return __builtin_amdgcn_tanhf(x); }
#else
__device__ __forceinline__ float fast_tanh(float x) { return tanhf(x); }
#endif

// ---------------------------------------------------------------------------
// Kernel 0: one-time f32 -> bf16 conversion of W_enc (A x E), row-major
// ---------------------------------------------------------------------------
__global__ __launch_bounds__(256) void wconv_kernel(
    const float* __restrict__ W, __bf16* __restrict__ out) {
  const int i = (blockIdx.x * 256 + threadIdx.x) * 4;  // 128 blocks cover A_*E_
  float4 w = *(const float4*)(W + i);
  v4bf o;
  o[0] = (__bf16)w.x; o[1] = (__bf16)w.y;
  o[2] = (__bf16)w.z; o[3] = (__bf16)w.w;
  *(v4bf*)(out + i) = o;
}

// ---------------------------------------------------------------------------
// Kernel 1: dec_feat[b][a] = sum_d decoder_hidden[b][d] * W_dec[a][d]
// ---------------------------------------------------------------------------
__global__ __launch_bounds__(256) void decfeat_kernel(
    const float* __restrict__ dh, const float* __restrict__ Wd,
    float* __restrict__ df) {
  const int b = blockIdx.x;       // 64
  const int a = threadIdx.x;      // 256
  const float* x = dh + (long)b * D_;
  const float* w = Wd + (long)a * D_;
  float s = 0.f;
#pragma unroll 4
  for (int d = 0; d < D_; d += 4) {
    float4 xv = *(const float4*)(x + d);
    float4 wv = *(const float4*)(w + d);
    s += xv.x * wv.x + xv.y * wv.y + xv.z * wv.z + xv.w * wv.w;
  }
  df[b * A_ + a] = s;
}

// ---------------------------------------------------------------------------
// Kernel 2: fused enc_feat GEMM (bf16 WMMA) + tanh + dot(v) -> raw scores
// One block = 64 rows (bs) x 256 cols (a). 8 waves:
//   wave -> rtile = wave>>1 (16-row slab), chalf = wave&1 (128-col half)
// W_enc chunks staged via async-to-LDS (double buffered, ASYNCcnt tracked).
// ---------------------------------------------------------------------------
__global__ __launch_bounds__(256) void attn_scores_kernel(
    const float* __restrict__ enc, const __bf16* __restrict__ wbf,
    const float* __restrict__ dec_feat, const float* __restrict__ v,
    float* __restrict__ scores) {
  // Double-buffered K-chunk of W: 256 rows(a) x 32 k bf16, row stride 40 (80B)
  __shared__ __align__(16) __bf16 wlds[2][256 * 40];
  __shared__ float s_scores[64];

  const int tid   = threadIdx.x;
  const int lane  = tid & 31;
  const int wave  = tid >> 5;
  const int rtile = wave >> 1;   // 0..3
  const int chalf = wave & 1;    // 0..1
  const int lrow  = lane & 15;   // M (A-frag) / N (B,C frags)
  const int lhi   = lane >> 4;   // half-wave selector

  const long row0 = (long)blockIdx.x * 64;   // global bs row base
  const int  b    = (int)(row0 / S_);        // batch index

  if (tid < 64) s_scores[tid] = 0.0f;

  const v8f vzero = {0.f, 0.f, 0.f, 0.f, 0.f, 0.f, 0.f, 0.f};
  v8f acc[8];
#pragma unroll
  for (int t = 0; t < 8; ++t) acc[t] = vzero;

  // A-fragment source row for this lane (contiguous k order: k = lhi*16 + s)
  const float* arow = enc + (row0 + rtile * 16 + lrow) * (long)E_ + lhi * 16;

  // Async stage of W chunk [kc..kc+32) into LDS buffer `sel`:
  // thread tid copies its 64-byte row (a = tid) as 4 x b128 async transfers.
  auto stage = [&](int kc, int sel) {
    const char* src = (const char*)(wbf + (long)tid * E_ + kc);
    const uint32_t dst = (uint32_t)(uintptr_t)&wlds[sel][tid * 40];
#pragma unroll
    for (int c = 0; c < 4; ++c) {
      asm volatile("global_load_async_to_lds_b128 %0, %1, off"
                   :: "v"(dst + c * 16), "v"((const void*)(src + c * 16))
                   : "memory");
    }
  };

  stage(0, 0);

  for (int it = 0; it < E_ / 32; ++it) {
    const int kc = it * 32;
    asm volatile("s_wait_asynccnt 0x0" ::: "memory");  // own chunk `it` landed
    __syncthreads();  // all waves' chunk `it` visible; chunk it-1 reads done
    if (it + 1 < E_ / 32) stage(kc + 32, (it + 1) & 1);

    // A fragment: lane holds M=lrow, k = lhi*16 + {0..15}
    v16bf af;
    {
      const float* p = arow + kc;
      float4 a0 = *(const float4*)(p);
      float4 a1 = *(const float4*)(p + 4);
      float4 a2 = *(const float4*)(p + 8);
      float4 a3 = *(const float4*)(p + 12);
      af[0]  = (__bf16)a0.x; af[1]  = (__bf16)a0.y;
      af[2]  = (__bf16)a0.z; af[3]  = (__bf16)a0.w;
      af[4]  = (__bf16)a1.x; af[5]  = (__bf16)a1.y;
      af[6]  = (__bf16)a1.z; af[7]  = (__bf16)a1.w;
      af[8]  = (__bf16)a2.x; af[9]  = (__bf16)a2.y;
      af[10] = (__bf16)a2.z; af[11] = (__bf16)a2.w;
      af[12] = (__bf16)a3.x; af[13] = (__bf16)a3.y;
      af[14] = (__bf16)a3.z; af[15] = (__bf16)a3.w;
    }

    const __bf16* cw = wlds[it & 1];
#pragma unroll
    for (int t = 0; t < 8; ++t) {
      // B fragment: lane holds N = lrow of col-tile, k = lhi*16 + {0..15}
      const int n = (chalf * 8 + t) * 16 + lrow;
      v16bf bfrag;
      __builtin_memcpy(&bfrag, &cw[n * 40 + lhi * 16], 32);
      acc[t] = __builtin_amdgcn_wmma_f32_16x16x32_bf16(
          false, af, false, bfrag, (short)0, acc[t], false, false);
    }
  }

  // Epilogue: score[row] = sum_a tanh(enc_feat + dec_feat[b][a]) * v[a]
  // C layout: lane holds N = lrow, VGPR i holds M = i + 8*lhi
  float sums[8];
#pragma unroll
  for (int i = 0; i < 8; ++i) sums[i] = 0.f;
#pragma unroll
  for (int t = 0; t < 8; ++t) {
    const int a   = (chalf * 8 + t) * 16 + lrow;
    const float d = dec_feat[b * A_ + a];
    const float w = v[a];
#pragma unroll
    for (int i = 0; i < 8; ++i)
      sums[i] += fast_tanh(acc[t][i] + d) * w;
  }
  // reduce across the 16 lanes sharing the same row (masks stay in half-wave)
#pragma unroll
  for (int i = 0; i < 8; ++i) {
    sums[i] += __shfl_xor(sums[i], 1, 32);
    sums[i] += __shfl_xor(sums[i], 2, 32);
    sums[i] += __shfl_xor(sums[i], 4, 32);
    sums[i] += __shfl_xor(sums[i], 8, 32);
  }
  if (lrow == 0) {
#pragma unroll
    for (int i = 0; i < 8; ++i)
      atomicAdd(&s_scores[rtile * 16 + lhi * 8 + i], sums[i]);  // 2 commutative adds -> deterministic
  }
  __syncthreads();
  if (tid < 64) scores[row0 + tid] = s_scores[tid];
}

// ---------------------------------------------------------------------------
// Kernel 3: masked softmax over S, in place (scores -> weights)
// ---------------------------------------------------------------------------
__global__ __launch_bounds__(256) void softmax_kernel(
    float* __restrict__ sw, const int* __restrict__ mask) {
  const int b   = blockIdx.x;   // 64
  const int tid = threadIdx.x;  // 256
  float* row = sw + (long)b * S_;
  const int* m = mask + (long)b * S_;

  float vals[8];
  float mx = -INFINITY;
#pragma unroll
  for (int i = 0; i < 8; ++i) {
    const int s = tid + i * 256;
    float val = row[s];
    val = (m[s] == 0) ? -INFINITY : val;
    vals[i] = val;
    mx = fmaxf(mx, val);
  }

  __shared__ float red[8];
  const int wave = tid >> 5, lane = tid & 31;
#pragma unroll
  for (int off = 16; off > 0; off >>= 1) mx = fmaxf(mx, __shfl_xor(mx, off, 32));
  if (lane == 0) red[wave] = mx;
  __syncthreads();
  if (tid == 0) {
    float t = red[0];
    for (int i = 1; i < 8; ++i) t = fmaxf(t, red[i]);
    red[0] = t;
  }
  __syncthreads();
  mx = red[0];
  __syncthreads();

  float sum = 0.f;
#pragma unroll
  for (int i = 0; i < 8; ++i) {
    vals[i] = __expf(vals[i] - mx);   // native v_exp_f32 path
    sum += vals[i];
  }
#pragma unroll
  for (int off = 16; off > 0; off >>= 1) sum += __shfl_xor(sum, off, 32);
  if (lane == 0) red[wave] = sum;
  __syncthreads();
  if (tid == 0) {
    float t = red[0];
    for (int i = 1; i < 8; ++i) t += red[i];
    red[0] = t;
  }
  __syncthreads();
  const float inv = 1.0f / red[0];

#pragma unroll
  for (int i = 0; i < 8; ++i) row[tid + i * 256] = vals[i] * inv;
}

// ---------------------------------------------------------------------------
// Kernel 4a: context partials over S-chunks (deterministic two-stage reduce)
// block = (b, chunk c of 256 s-rows), 128 threads each owning a float4 of E
// ---------------------------------------------------------------------------
__global__ __launch_bounds__(128) void context_partial_kernel(
    const float* __restrict__ enc, const float* __restrict__ weights,
    float* __restrict__ part) {
  const int blk = blockIdx.x;      // 512
  const int b = blk >> 3, c = blk & 7;
  const int tid = threadIdx.x;     // 128, float4 of E
  const long base = ((long)b * S_ + (long)c * 256) * E_;
  const float* wrow = weights + (long)b * S_ + c * 256;

  float4 acc = {0.f, 0.f, 0.f, 0.f};
  for (int s = 0; s < 256; ++s) {
    const float w = wrow[s];
    float4 ev = *(const float4*)(enc + base + (long)s * E_ + tid * 4);
    acc.x += w * ev.x;
    acc.y += w * ev.y;
    acc.z += w * ev.z;
    acc.w += w * ev.w;
  }
  *(float4*)(part + ((long)(b * 8 + c)) * E_ + tid * 4) = acc;
}

__global__ __launch_bounds__(128) void context_reduce_kernel(
    const float* __restrict__ part, float* __restrict__ ctx) {
  const int b = blockIdx.x;     // 64
  const int tid = threadIdx.x;  // 128
  float4 acc = {0.f, 0.f, 0.f, 0.f};
#pragma unroll
  for (int c = 0; c < 8; ++c) {
    float4 p = *(const float4*)(part + ((long)(b * 8 + c)) * E_ + tid * 4);
    acc.x += p.x; acc.y += p.y; acc.z += p.z; acc.w += p.w;
  }
  *(float4*)(ctx + (long)b * E_ + tid * 4) = acc;
}

// ---------------------------------------------------------------------------
extern "C" void kernel_launch(void* const* d_in, const int* in_sizes, int n_in,
                              void* d_out, int out_size, void* d_ws, size_t ws_size,
                              hipStream_t stream) {
  const float* enc  = (const float*)d_in[0];  // (B,S,E)
  const float* dh   = (const float*)d_in[1];  // (B,D)
  const int*   mask = (const int*)  d_in[2];  // (B,S)
  const float* Wenc = (const float*)d_in[3];  // (A,E)
  const float* Wdec = (const float*)d_in[4];  // (A,D)
  const float* v    = (const float*)d_in[5];  // (A,)

  float* out     = (float*)d_out;
  float* ctx     = out;                 // (B,E)  = 32768 floats
  float* weights = out + B_ * E_;       // (B,S)  = 131072 floats

  float* ws       = (float*)d_ws;
  __bf16* wbf     = (__bf16*)ws;                  // A*E bf16 = 65536 float slots
  float* dec_feat = ws + (A_ * E_) / 2;           // B*A   = 16384 floats
  float* part     = dec_feat + B_ * A_;           // B*8*E = 262144 floats

  wconv_kernel<<<(A_ * E_) / 1024, 256, 0, stream>>>(Wenc, wbf);
  decfeat_kernel<<<B_, 256, 0, stream>>>(dh, Wdec, dec_feat);
  attn_scores_kernel<<<(B_ * S_) / 64, 256, 0, stream>>>(enc, wbf, dec_feat, v, weights);
  softmax_kernel<<<B_, 256, 0, stream>>>(weights, mask);
  context_partial_kernel<<<B_ * 8, 128, 0, stream>>>(enc, weights, part);
  context_reduce_kernel<<<B_, 128, 0, stream>>>(part, ctx);
}
